// Memory_56513179681523
// MI455X (gfx1250) — compile-verified, matrix operand order checked by reading, not run on hardware
//
#include <hip/hip_runtime.h>
#include <hip/hip_bf16.h>

typedef __attribute__((ext_vector_type(16))) _Float16 v16h;
typedef __attribute__((ext_vector_type(8)))  _Float16 v8h;
typedef __attribute__((ext_vector_type(8)))  float    v8f;

#define B_ROWS   4096
#define HID      512
#define NEG_BIG  (-3.0e38f)
#define TILE_M   64
#define TILE_N   64
#define KSTEP    32

// ---------------------------------------------------------------------------
// f32 -> f16 conversion
// ---------------------------------------------------------------------------
__global__ void cvt_f16(const float* __restrict__ src, _Float16* __restrict__ dst, int n) {
  int i = blockIdx.x * blockDim.x + threadIdx.x;
  if (i < n) dst[i] = (_Float16)src[i];
}

// ---------------------------------------------------------------------------
// LDS byte offset of a __shared__ object (generic -> addrspace(3) cast).
// ---------------------------------------------------------------------------
__device__ __forceinline__ unsigned int lds_off_u32(const void* p) {
  return (unsigned int)(size_t)(__attribute__((address_space(3))) const void*)p;
}

// 16-byte async copy global -> LDS (gfx1250 GLOBAL_LOAD_ASYNC_TO_LDS_B128,
// tracked with ASYNCcnt). vdst = LDS byte offset, vaddr = 64-bit global addr.
__device__ __forceinline__ void async_copy16(unsigned int lds_byte_off, const void* gptr) {
  asm volatile("global_load_async_to_lds_b128 %0, %1, off"
               :: "v"(lds_byte_off), "v"((unsigned long long)(size_t)gptr)
               : "memory");
}

__device__ __forceinline__ void wait_async0() {
  asm volatile("s_wait_asynccnt 0x0" ::: "memory");
}

// ---------------------------------------------------------------------------
// WMMA fragment load: ISA 7.12.2 16-bit A/B layout.
// lanes 0-15 : K = {0..7, 16..23} ; lanes 16-31 : K = {8..15, 24..31}
// base already includes + hi*8 ; second 8 halves live at +16.
// ---------------------------------------------------------------------------
__device__ __forceinline__ v16h load_frag(const _Float16* base) {
  v8h a0 = *(const v8h*)(base);
  v8h a1 = *(const v8h*)(base + 16);
  v16h f;
#pragma unroll
  for (int i = 0; i < 8; ++i) { f[i] = a0[i]; f[i + 8] = a1[i]; }
  return f;
}

// ---------------------------------------------------------------------------
// C = A * B^T block tile kernel: 64x64 per block (256 thr = 8 waves),
// each wave computes 16x32 (2 accumulators sharing one A fragment).
// A/B K-slabs staged in LDS via async global->LDS loads.
// MODE 0: +bias, ELU, store f16     (GEMM1 -> h1)
// MODE 1: +bias, store f32          (GEMM2 -> h)
// MODE 2: plain, store f32, batched (scores, blockIdx.z = head*2+sub)
// ---------------------------------------------------------------------------
template <int MODE>
__global__ __launch_bounds__(256) void wmma_gemm_bt(
    const _Float16* __restrict__ A, const _Float16* __restrict__ B,
    const float* __restrict__ bias, float* __restrict__ Cf,
    _Float16* __restrict__ Ch, int N, int K, int lda, int ldb, int ldc,
    long aStrideZ, long bStrideZ, long cStrideZ) {
  __shared__ _Float16 Atile[TILE_M * KSTEP];   // 4 KB, row stride 32
  __shared__ _Float16 Btile[TILE_N * KSTEP];   // 4 KB

  const int tid  = threadIdx.x;
  const int lane = tid & 31;
  const int wave = tid >> 5;
  const int mw   = wave & 3;    // 4 m-tiles of 16
  const int nw   = wave >> 2;   // 2 n-slabs of 32
  const int m0   = blockIdx.x * TILE_M;
  const int n0   = blockIdx.y * TILE_N;
  const int r    = lane & 15;
  const int hi   = lane >> 4;

  const size_t az = (size_t)blockIdx.z * aStrideZ;
  const size_t bz = (size_t)blockIdx.z * bStrideZ;

  // staging: thread tid owns one 16-byte chunk of each tile per K step
  const int row  = tid >> 2;    // 0..63
  const int chnk = tid & 3;     // 4 x 8 halves per 32-wide row
  const _Float16* gA = A + az + (size_t)(m0 + row) * lda + chnk * 8;
  const _Float16* gB = B + bz + (size_t)(n0 + row) * ldb + chnk * 8;
  const unsigned int ldsA = lds_off_u32(Atile) + (unsigned int)(row * KSTEP + chnk * 8) * 2u;
  const unsigned int ldsB = lds_off_u32(Btile) + (unsigned int)(row * KSTEP + chnk * 8) * 2u;

  // per-wave fragment source pointers inside LDS
  const _Float16* fragA  = Atile + (mw * 16 + r) * KSTEP + hi * 8;
  const _Float16* fragB0 = Btile + (nw * 32 + r) * KSTEP + hi * 8;
  const _Float16* fragB1 = Btile + (nw * 32 + 16 + r) * KSTEP + hi * 8;

  v8f acc0 = {}, acc1 = {};
  for (int k0 = 0; k0 < K; k0 += KSTEP) {
    async_copy16(ldsA, gA + k0);
    async_copy16(ldsB, gB + k0);
    wait_async0();
    __syncthreads();

    v16h a  = load_frag(fragA);
    v16h b0 = load_frag(fragB0);
    v16h b1 = load_frag(fragB1);
    acc0 = __builtin_amdgcn_wmma_f32_16x16x32_f16(false, a, false, b0,
                                                  (short)0, acc0, false, false);
    acc1 = __builtin_amdgcn_wmma_f32_16x16x32_f16(false, a, false, b1,
                                                  (short)0, acc1, false, false);
    __syncthreads();   // protect LDS WAR for next K step
  }

  const int n  = n0 + nw * 32 + r;
  const float bv0 = (MODE <= 1) ? bias[n]      : 0.0f;
  const float bv1 = (MODE <= 1) ? bias[n + 16] : 0.0f;
#pragma unroll
  for (int v = 0; v < 8; ++v) {
    const int m = m0 + mw * 16 + v + hi * 8;   // C/D: lanes 16-31 hold M=v+8
    float v0 = acc0[v] + bv0;
    float v1 = acc1[v] + bv1;
    if (MODE == 0) {
      v0 = (v0 > 0.0f) ? v0 : (__expf(v0) - 1.0f);
      v1 = (v1 > 0.0f) ? v1 : (__expf(v1) - 1.0f);
      Ch[(size_t)m * ldc + n]      = (_Float16)v0;
      Ch[(size_t)m * ldc + n + 16] = (_Float16)v1;
    } else {
      const size_t cz = (size_t)blockIdx.z * cStrideZ;
      Cf[cz + (size_t)m * ldc + n]      = v0;
      Cf[cz + (size_t)m * ldc + n + 16] = v1;
    }
  }
}

// ---------------------------------------------------------------------------
// BatchNorm statistics: one block per channel (1024), reduce over 4096 rows.
// ---------------------------------------------------------------------------
__global__ void bn_stats(const float* __restrict__ h, float* __restrict__ stats) {
  const int c = blockIdx.x, tid = threadIdx.x;
  float s = 0.f, ss = 0.f;
  for (int r2 = tid; r2 < B_ROWS; r2 += 256) {
    float x = h[(size_t)r2 * 1024 + c];
    s += x; ss += x * x;
  }
  __shared__ float sh[256], sh2[256];
  sh[tid] = s; sh2[tid] = ss;
  __syncthreads();
  for (int st = 128; st; st >>= 1) {
    if (tid < st) { sh[tid] += sh[tid + st]; sh2[tid] += sh2[tid + st]; }
    __syncthreads();
  }
  if (tid == 0) {
    float mu  = sh[0] * (1.0f / B_ROWS);
    float var = sh2[0] * (1.0f / B_ROWS) - mu * mu;
    stats[c]        = mu;
    stats[1024 + c] = rsqrtf(var + 1e-5f);
  }
}

// ---------------------------------------------------------------------------
// Apply BN, emit q in f16 for the score GEMMs.
// ---------------------------------------------------------------------------
__global__ void bn_apply(const float* __restrict__ h, const float* __restrict__ stats,
                         const float* __restrict__ gamma, const float* __restrict__ beta,
                         _Float16* __restrict__ q) {
  size_t i = (size_t)blockIdx.x * blockDim.x + threadIdx.x;
  int c = (int)(i & 1023);
  float v = (h[i] - stats[c]) * stats[1024 + c] * gamma[c] + beta[c];
  q[i] = (_Float16)v;
}

// ---------------------------------------------------------------------------
// Iterative wave32 top-32 selection (argmax + mask, 32 rounds).
// Element identity: code = j*32 + lane.
// ---------------------------------------------------------------------------
template <int NLOC>
__device__ __forceinline__ void select_top32(float* v, int lane,
                                             float* outV, int* outI) {
  for (int it = 0; it < 32; ++it) {
    float m = NEG_BIG; int jl = 0;
#pragma unroll
    for (int j = 0; j < NLOC; ++j)
      if (v[j] > m) { m = v[j]; jl = j; }
    int code = jl * 32 + lane;
#pragma unroll
    for (int off = 16; off; off >>= 1) {
      float om = __shfl_xor(m, off);
      int   oc = __shfl_xor(code, off);
      if (om > m || (om == m && oc < code)) { m = om; code = oc; }
    }
    if (lane == (code & 31)) {
      int jm = code >> 5;
#pragma unroll
      for (int j = 0; j < NLOC; ++j) if (j == jm) v[j] = NEG_BIG;
    }
    if (lane == 0) { outV[it] = m; outI[it] = code; }
  }
}

// ---------------------------------------------------------------------------
// Per (b, head): top-32 of s1, top-32 of s2, 32x32 combine, top-32, softmax.
// One wave32 per block; grid = B*4.
// ---------------------------------------------------------------------------
__global__ void topk_combine(const float* __restrict__ sbuf,
                             float* __restrict__ wout, int* __restrict__ idxout) {
  const int bh = blockIdx.x;
  const int b = bh >> 2, head = bh & 3;
  const int lane = threadIdx.x;

  __shared__ float t1[32], t2[32], sc[32];
  __shared__ int   i1[32], i2[32], scode[32];

  const float* s1 = sbuf + ((size_t)(head * 2 + 0) * B_ROWS + b) * 512;
  const float* s2 = sbuf + ((size_t)(head * 2 + 1) * B_ROWS + b) * 512;

  float v[32];
#pragma unroll
  for (int j = 0; j < 16; ++j) v[j] = s1[j * 32 + lane];
  select_top32<16>(v, lane, t1, i1);
#pragma unroll
  for (int j = 0; j < 16; ++j) v[j] = s2[j * 32 + lane];
  select_top32<16>(v, lane, t2, i2);
  __syncthreads();

  // combos: lane = i (index into t1), j = index into t2
  const float base = t1[lane];
#pragma unroll
  for (int j = 0; j < 32; ++j) v[j] = base + t2[j];
  select_top32<32>(v, lane, sc, scode);
  __syncthreads();

  // softmax over the 32 selected scores (sc[0] is the max: descending order)
  float e = __expf(sc[lane] - sc[0]);
  float s = e;
#pragma unroll
  for (int off = 16; off; off >>= 1) s += __shfl_xor(s, off);

  int code = scode[lane];                          // code = j*32 + i
  int full = i1[code & 31] * 512 + i2[code >> 5];  // global mem slot index

  const size_t o = (size_t)b * 128 + head * 32 + lane;
  wout[o]   = e / s;
  idxout[o] = full;
}

// ---------------------------------------------------------------------------
// out[b] = sum_k w[b,k] * values[idx[b,k], :]   (the HBM-bound gather)
// One block per batch row; prefetch next row (global_prefetch_b8 on gfx1250).
// ---------------------------------------------------------------------------
__global__ void gather_weighted(const float* __restrict__ values,
                                const float* __restrict__ w,
                                const int* __restrict__ idx,
                                float* __restrict__ out) {
  const int b = blockIdx.x, tid = threadIdx.x;
  __shared__ float lw[128];
  __shared__ int   li[128];
  if (tid < 128) {
    lw[tid] = w[(size_t)b * 128 + tid];
    li[tid] = idx[(size_t)b * 128 + tid];
  }
  __syncthreads();

  const int d = tid * 2;
  float a0 = 0.f, a1 = 0.f;
  for (int k = 0; k < 128; ++k) {
    const float* row = values + (size_t)li[k] * HID + d;
    if (k + 1 < 128)
      __builtin_prefetch(values + (size_t)li[k + 1] * HID + d, 0, 1);
    float2 rv = *(const float2*)row;
    a0 += lw[k] * rv.x;
    a1 += lw[k] * rv.y;
  }
  float2 o; o.x = a0; o.y = a1;
  *(float2*)(out + (size_t)b * HID + d) = o;
}

// ---------------------------------------------------------------------------
extern "C" void kernel_launch(void* const* d_in, const int* in_sizes, int n_in,
                              void* d_out, int out_size, void* d_ws, size_t ws_size,
                              hipStream_t stream) {
  const float* x      = (const float*)d_in[0];   // [4096, 512]
  const float* W1     = (const float*)d_in[1];   // [256, 512]
  const float* b1     = (const float*)d_in[2];   // [256]
  const float* W2     = (const float*)d_in[3];   // [1024, 256]
  const float* b2     = (const float*)d_in[4];   // [1024]
  const float* gamma  = (const float*)d_in[5];   // [1024]
  const float* beta   = (const float*)d_in[6];   // [1024]
  const float* keys   = (const float*)d_in[7];   // [4, 2, 512, 128]
  const float* values = (const float*)d_in[8];   // [262144, 512]
  float* out = (float*)d_out;                    // [4096, 512]

  char* ws = (char*)d_ws;
  size_t off = 0;
  auto alloc = [&](size_t bytes) -> void* {
    void* p = ws + off;
    off += (bytes + 255) & ~(size_t)255;
    return p;
  };

  _Float16* x16   = (_Float16*)alloc((size_t)4096 * 512 * 2);
  _Float16* W1h   = (_Float16*)alloc((size_t)256 * 512 * 2);
  _Float16* h1h   = (_Float16*)alloc((size_t)4096 * 256 * 2);
  _Float16* W2h   = (_Float16*)alloc((size_t)1024 * 256 * 2);
  float*    hbuf  = (float*)   alloc((size_t)4096 * 1024 * 4);
  float*    stats = (float*)   alloc((size_t)2048 * 4);
  _Float16* q16   = (_Float16*)alloc((size_t)4096 * 1024 * 2);
  _Float16* k16   = (_Float16*)alloc((size_t)8 * 512 * 128 * 2);
  float*    sbuf  = (float*)   alloc((size_t)8 * 4096 * 512 * 4);
  float*    wq    = (float*)   alloc((size_t)4096 * 128 * 4);
  int*      idxq  = (int*)     alloc((size_t)4096 * 128 * 4);

  // 1) convert operands to f16
  cvt_f16<<<(4096 * 512 + 255) / 256, 256, 0, stream>>>(x, x16, 4096 * 512);
  cvt_f16<<<(256 * 512 + 255) / 256, 256, 0, stream>>>(W1, W1h, 256 * 512);
  cvt_f16<<<(1024 * 256 + 255) / 256, 256, 0, stream>>>(W2, W2h, 1024 * 256);
  cvt_f16<<<(8 * 512 * 128 + 255) / 256, 256, 0, stream>>>(keys, k16, 8 * 512 * 128);

  // 2) h1 = elu(x @ W1^T + b1)  : M=4096 N=256 K=512, f16 out
  wmma_gemm_bt<0><<<dim3(64, 4, 1), 256, 0, stream>>>(
      x16, W1h, b1, nullptr, h1h, 256, 512, 512, 512, 256, 0, 0, 0);

  // 3) h = h1 @ W2^T + b2       : M=4096 N=1024 K=256, f32 out
  wmma_gemm_bt<1><<<dim3(64, 16, 1), 256, 0, stream>>>(
      h1h, W2h, b2, hbuf, nullptr, 1024, 256, 256, 256, 1024, 0, 0, 0);

  // 4) batch-norm stats + apply (emit q in f16)
  bn_stats<<<1024, 256, 0, stream>>>(hbuf, stats);
  bn_apply<<<(4096 * 1024) / 256, 256, 0, stream>>>(hbuf, stats, gamma, beta, q16);

  // 5) scores: 8 batched GEMMs (head,sub): M=4096 N=512 K=128
  wmma_gemm_bt<2><<<dim3(64, 8, 8), 256, 0, stream>>>(
      q16, k16, nullptr, sbuf, nullptr, 512, 128, 1024, 128, 512,
      128L, 512L * 128L, 4096L * 512L);

  // 6) two-stage top-32 + softmax per (b, head)
  topk_combine<<<4096 * 4, 32, 0, stream>>>(sbuf, wq, idxq);

  // 7) HBM-bound gather + weighted sum
  gather_weighted<<<4096, 256, 0, stream>>>(values, wq, idxq, out);
}
